// DotProductAttention_19473381720281
// MI455X (gfx1250) — compile-verified, matrix operand order checked by reading, not run on hardware
//
#include <hip/hip_runtime.h>
#include <stdint.h>

// ---------------- CDNA5 WMMA types ----------------
typedef __attribute__((ext_vector_type(16))) __bf16 v16bf;
typedef __attribute__((ext_vector_type(8)))  float  v8f;

#define DEV static __device__ __forceinline__

union FragBF { v16bf v; uint4 q[2]; uint16_t s[16]; };

DEV v8f wmma_bf16(FragBF a, FragBF b, v8f c) {
  return __builtin_amdgcn_wmma_f32_16x16x32_bf16(
      /*neg_a=*/false, a.v, /*neg_b=*/false, b.v,
      /*c_mod=*/(short)0, c, /*reuse_a=*/false, /*reuse_b=*/false);
}

DEV v8f v8f_zero() {
  v8f z;
  #pragma unroll
  for (int i = 0; i < 8; ++i) z[i] = 0.f;
  return z;
}

DEV uint16_t f2bf(float x) {               // RNE f32 -> bf16
  union { float f; uint32_t u; } v; v.f = x;
  uint32_t r = v.u + 0x7FFFu + ((v.u >> 16) & 1u);
  return (uint16_t)(r >> 16);
}

// ---------------- problem constants ----------------
constexpr int Bc = 8, Fc = 16, Pc = 196, Dc = 768, H2c = 6;
constexpr int Nc  = Fc * Pc;        // 3136 tokens per batch
constexpr int Mc  = Bc * Nc;        // 25088 total rows
constexpr int E3c = 3 * Dc;         // 2304 qkv cols

// workspace layout (bytes, all 256B aligned)
constexpr size_t XB_BYTES    = (size_t)Mc  * Dc  * 2;   // x bf16
constexpr size_t WQKV_BYTES  = (size_t)E3c * Dc  * 2;   // w_qkv bf16
constexpr size_t WPROJ_BYTES = (size_t)Dc  * Dc  * 2;   // w_proj bf16
constexpr size_t QKV_BYTES   = (size_t)Mc  * E3c * 2;   // qkv bf16
constexpr size_t XB_OFF = 0;
constexpr size_t WQKV_OFF  = XB_OFF    + XB_BYTES;
constexpr size_t WPROJ_OFF = WQKV_OFF  + WQKV_BYTES;
constexpr size_t QKV_OFF   = WPROJ_OFF + WPROJ_BYTES;
constexpr size_t ATTN_OFF  = QKV_OFF   + QKV_BYTES;

// ---------------- f32 -> bf16 convert ----------------
__global__ __launch_bounds__(256)
void cvt_f32_bf16(const float* __restrict__ in, uint16_t* __restrict__ out, int n) {
  int i = blockIdx.x * 256 + threadIdx.x;
  if (i < n) out[i] = f2bf(in[i]);
}

// ---------------- GEMM: C[M,N] = A[M,K] * B[N,K]^T ----------------
// block = 256 threads = 8 waves; block tile 128x128, K-step 32, double-buffered
// LDS. wave (wm in 0..3, wn in 0..1) computes 32x64 via 2x4 WMMA accumulators.
// Global loads for step k+1 are staged in registers while WMMAs consume step k;
// one barrier per K-step. global_prefetch pulls step k+2 toward the caches.
template <bool F32OUT>
__global__ __launch_bounds__(256)
void gemm_bf16(const uint16_t* __restrict__ A, const uint16_t* __restrict__ Bm,
               uint16_t* __restrict__ Cb, float* __restrict__ Cf,
               const float* __restrict__ bias, int K, int Ncols) {
  __shared__ uint16_t As[2][128 * 40];  // pitch 40 halves (80B, 16B aligned)
  __shared__ uint16_t Bs[2][128 * 40];

  const int tid = threadIdx.x, lane = tid & 31, wave = tid >> 5;
  const int wm = wave >> 1, wn = wave & 1;
  const int mBase = blockIdx.y * 128, nBase = blockIdx.x * 128;

  v8f c[2][4];
  #pragma unroll
  for (int i = 0; i < 2; ++i)
    #pragma unroll
    for (int j = 0; j < 4; ++j) c[i][j] = v8f_zero();

  const int lr = tid >> 1, lc = (tid & 1) * 16;   // LDS staging: 32B per thread
  const int koff = (lane & 16) ? 8 : 0;

  const uint16_t* ga = A  + (size_t)(mBase + lr) * K + lc;  // this thread's row
  const uint16_t* gb = Bm + (size_t)(nBase + lr) * K + lc;

  // prime buffer 0
  {
    uint4 ra0 = *(const uint4*)ga,       ra1 = *(const uint4*)(ga + 8);
    uint4 rb0 = *(const uint4*)gb,       rb1 = *(const uint4*)(gb + 8);
    *(uint4*)&As[0][lr * 40 + lc]     = ra0;
    *(uint4*)&As[0][lr * 40 + lc + 8] = ra1;
    *(uint4*)&Bs[0][lr * 40 + lc]     = rb0;
    *(uint4*)&Bs[0][lr * 40 + lc + 8] = rb1;
  }
  __syncthreads();

  const int nk = K / 32;
  for (int ks = 0; ks < nk; ++ks) {
    const int cur = ks & 1, nxt = cur ^ 1;

    uint4 ra0, ra1, rb0, rb1;
    const bool have_next = (ks + 1 < nk);
    if (have_next) {                    // stage k+1 while computing k
      const uint16_t* na = ga + (ks + 1) * 32;
      const uint16_t* nb = gb + (ks + 1) * 32;
      ra0 = *(const uint4*)na; ra1 = *(const uint4*)(na + 8);
      rb0 = *(const uint4*)nb; rb1 = *(const uint4*)(nb + 8);
      if (ks + 2 < nk) {                // global_prefetch_b8 for k+2
        __builtin_prefetch(ga + (ks + 2) * 32, 0, 0);
        __builtin_prefetch(gb + (ks + 2) * 32, 0, 0);
      }
    }

    FragBF a[2], b[4];
    #pragma unroll
    for (int mt = 0; mt < 2; ++mt) {
      const int row = wm * 32 + mt * 16 + (lane & 15);
      a[mt].q[0] = *(const uint4*)&As[cur][row * 40 + koff];
      a[mt].q[1] = *(const uint4*)&As[cur][row * 40 + 16 + koff];
    }
    #pragma unroll
    for (int nt = 0; nt < 4; ++nt) {
      const int row = wn * 64 + nt * 16 + (lane & 15);
      b[nt].q[0] = *(const uint4*)&Bs[cur][row * 40 + koff];
      b[nt].q[1] = *(const uint4*)&Bs[cur][row * 40 + 16 + koff];
    }
    #pragma unroll
    for (int mt = 0; mt < 2; ++mt)
      #pragma unroll
      for (int nt = 0; nt < 4; ++nt)
        c[mt][nt] = wmma_bf16(a[mt], b[nt], c[mt][nt]);

    if (have_next) {
      *(uint4*)&As[nxt][lr * 40 + lc]     = ra0;
      *(uint4*)&As[nxt][lr * 40 + lc + 8] = ra1;
      *(uint4*)&Bs[nxt][lr * 40 + lc]     = rb0;
      *(uint4*)&Bs[nxt][lr * 40 + lc + 8] = rb1;
    }
    __syncthreads();
  }

  const int mh = (lane & 16) ? 8 : 0;
  #pragma unroll
  for (int mt = 0; mt < 2; ++mt)
    #pragma unroll
    for (int nt = 0; nt < 4; ++nt)
      #pragma unroll
      for (int r = 0; r < 8; ++r) {
        const int m = mBase + wm * 32 + mt * 16 + r + mh;
        const int n = nBase + wn * 64 + nt * 16 + (lane & 15);
        const float v = c[mt][nt][r];
        if (F32OUT) Cf[(size_t)m * Ncols + n] = v + bias[n];
        else        Cb[(size_t)m * Ncols + n] = f2bf(v);
      }
}

// ---------------- spatial attention (heads 0..5, 196x196 per b,f,h) --------
// block = 256 = 8 waves; each wave = 16 query rows; 2 blocks cover 196 rows.
__global__ __launch_bounds__(256)
void attn_spatial(const uint16_t* __restrict__ qkv, uint16_t* __restrict__ attn) {
  __shared__ uint16_t Ks[Pc * 64];       // K row-major [196][64]
  __shared__ uint16_t Vt[64 * 224];      // V transposed [64][224], cols>=196 zero
  __shared__ uint16_t Ps[8][16 * 32];    // per-wave probability staging

  int t = blockIdx.x;
  const int qb = t & 1;  t >>= 1;
  const int h  = t % H2c; t /= H2c;
  const int f  = t % Fc;  const int b = t / Fc;

  const int tid = threadIdx.x, lane = tid & 31, wave = tid >> 5;
  const int rowTok0 = b * Nc + f * Pc;
  const int qcol = h * 64, kcol = Dc + h * 64, vcol = 2 * Dc + h * 64;

  // stage K (row-major) and V (transposed) into LDS
  for (int i = tid; i < Pc * 8; i += 256) {
    const int r = i >> 3, ch = i & 7;
    const size_t rbase = (size_t)(rowTok0 + r) * E3c;
    *(uint4*)&Ks[r * 64 + ch * 8] = *(const uint4*)&qkv[rbase + kcol + ch * 8];
    const uint4 vv = *(const uint4*)&qkv[rbase + vcol + ch * 8];
    const uint16_t* pv = (const uint16_t*)&vv;
    #pragma unroll
    for (int j = 0; j < 8; ++j) Vt[(ch * 8 + j) * 224 + r] = pv[j];
  }
  for (int i = tid; i < 64 * 28; i += 256)        // zero the K-padding cols
    Vt[(i / 28) * 224 + 196 + (i % 28)] = 0;
  __syncthreads();

  const int koff = (lane & 16) ? 8 : 0;
  const int mh   = (lane & 16) ? 8 : 0;
  const int q0 = qb * 128 + wave * 16;
  int qrow = q0 + (lane & 15); if (qrow > Pc - 1) qrow = Pc - 1;  // clamp pad rows

  FragBF aq[2];                                   // Q fragments, reused all tiles
  #pragma unroll
  for (int kk = 0; kk < 2; ++kk) {
    const uint16_t* g = &qkv[(size_t)(rowTok0 + qrow) * E3c + qcol + kk * 32 + koff];
    aq[kk].q[0] = *(const uint4*)g;
    aq[kk].q[1] = *(const uint4*)(g + 16);
  }

  float mrow[8], lrow[8]; v8f o[4];
  #pragma unroll
  for (int r = 0; r < 8; ++r) { mrow[r] = -1e30f; lrow[r] = 0.f; }
  #pragma unroll
  for (int nt = 0; nt < 4; ++nt) o[nt] = v8f_zero();

  const float scale = 0.125f;                     // 1/sqrt(64)
  uint16_t* ps = &Ps[wave][0];

  for (int jt = 0; jt < 7; ++jt) {                // 14 key tiles of 16 (pairs)
    float sarr[2][8];
    #pragma unroll
    for (int half = 0; half < 2; ++half) {
      const int j = jt * 2 + half;
      int krow = j * 16 + (lane & 15);
      const bool oob = (krow >= Pc); if (oob) krow = Pc - 1;
      FragBF bk0, bk1;
      bk0.q[0] = *(const uint4*)&Ks[krow * 64 + koff];
      bk0.q[1] = *(const uint4*)&Ks[krow * 64 + 16 + koff];
      bk1.q[0] = *(const uint4*)&Ks[krow * 64 + 32 + koff];
      bk1.q[1] = *(const uint4*)&Ks[krow * 64 + 48 + koff];
      v8f s = v8f_zero();
      s = wmma_bf16(aq[0], bk0, s);
      s = wmma_bf16(aq[1], bk1, s);
      #pragma unroll
      for (int r = 0; r < 8; ++r) sarr[half][r] = oob ? -1e30f : s[r] * scale;
    }
    // online softmax update (row reductions across 16-lane half-groups)
    #pragma unroll
    for (int r = 0; r < 8; ++r) {
      float v = fmaxf(sarr[0][r], sarr[1][r]);
      v = fmaxf(v, __shfl_xor(v, 1, 32)); v = fmaxf(v, __shfl_xor(v, 2, 32));
      v = fmaxf(v, __shfl_xor(v, 4, 32)); v = fmaxf(v, __shfl_xor(v, 8, 32));
      const float mnew = fmaxf(mrow[r], v);
      const float fac  = __expf(mrow[r] - mnew);
      const float e0 = __expf(sarr[0][r] - mnew);
      const float e1 = __expf(sarr[1][r] - mnew);
      float rs = e0 + e1;
      rs += __shfl_xor(rs, 1, 32); rs += __shfl_xor(rs, 2, 32);
      rs += __shfl_xor(rs, 4, 32); rs += __shfl_xor(rs, 8, 32);
      lrow[r] = lrow[r] * fac + rs;
      mrow[r] = mnew;
      #pragma unroll
      for (int nt = 0; nt < 4; ++nt) o[nt][r] *= fac;
      const int m = r + mh;
      ps[m * 32 +      (lane & 15)] = f2bf(e0);   // tile j even -> cols 0..15
      ps[m * 32 + 16 + (lane & 15)] = f2bf(e1);   // tile j odd  -> cols 16..31
    }
    asm volatile("s_wait_dscnt 0" ::: "memory");  // same-wave LDS turnaround
    FragBF pf;
    pf.q[0] = *(const uint4*)&ps[(lane & 15) * 32 + koff];
    pf.q[1] = *(const uint4*)&ps[(lane & 15) * 32 + 16 + koff];
    #pragma unroll
    for (int nt = 0; nt < 4; ++nt) {
      const int d = nt * 16 + (lane & 15);
      FragBF bv;
      bv.q[0] = *(const uint4*)&Vt[d * 224 + jt * 32 + koff];
      bv.q[1] = *(const uint4*)&Vt[d * 224 + jt * 32 + 16 + koff];
      o[nt] = wmma_bf16(pf, bv, o[nt]);
    }
    asm volatile("" ::: "memory");                // keep iterations ordered
  }

  #pragma unroll
  for (int r = 0; r < 8; ++r) {
    const int p = q0 + r + mh;
    if (p < Pc) {
      const float inv = 1.f / lrow[r];
      const size_t row = (size_t)(rowTok0 + p) * Dc;
      #pragma unroll
      for (int nt = 0; nt < 4; ++nt)
        attn[row + h * 64 + nt * 16 + (lane & 15)] = f2bf(o[nt][r] * inv);
    }
  }
}

// ---------------- temporal attention (heads 6..11, 16x16 per b,p,h) --------
__global__ __launch_bounds__(32)
void attn_temporal(const uint16_t* __restrict__ qkv, uint16_t* __restrict__ attn) {
  __shared__ uint16_t Ps[16 * 32];
  int t = blockIdx.x;
  const int h = t % H2c; t /= H2c;
  const int p = t % Pc;  const int b = t / Pc;
  const int lane = threadIdx.x & 31;
  const int gh = H2c + h;
  const int qcol = gh * 64, kcol = Dc + gh * 64, vcol = 2 * Dc + gh * 64;
  const int koff = (lane & 16) ? 8 : 0;
  const int mh   = (lane & 16) ? 8 : 0;

  const int fq = lane & 15;                      // query row / key col per lane
  const size_t tokq = (size_t)(b * Nc + fq * Pc + p);

  FragBF aq[2], bk[2];
  #pragma unroll
  for (int kk = 0; kk < 2; ++kk) {
    const uint16_t* g = &qkv[tokq * E3c + qcol + kk * 32 + koff];
    aq[kk].q[0] = *(const uint4*)g;  aq[kk].q[1] = *(const uint4*)(g + 16);
    g = &qkv[tokq * E3c + kcol + kk * 32 + koff];
    bk[kk].q[0] = *(const uint4*)g;  bk[kk].q[1] = *(const uint4*)(g + 16);
  }
  v8f s = v8f_zero();
  s = wmma_bf16(aq[0], bk[0], s);
  s = wmma_bf16(aq[1], bk[1], s);

  const float scale = 0.125f;
  #pragma unroll
  for (int r = 0; r < 8; ++r) {
    float sv = s[r] * scale;
    float mx = sv;
    mx = fmaxf(mx, __shfl_xor(mx, 1, 32)); mx = fmaxf(mx, __shfl_xor(mx, 2, 32));
    mx = fmaxf(mx, __shfl_xor(mx, 4, 32)); mx = fmaxf(mx, __shfl_xor(mx, 8, 32));
    float e = __expf(sv - mx);
    float rs = e;
    rs += __shfl_xor(rs, 1, 32); rs += __shfl_xor(rs, 2, 32);
    rs += __shfl_xor(rs, 4, 32); rs += __shfl_xor(rs, 8, 32);
    const int m = r + mh;
    Ps[m * 32 +      (lane & 15)] = f2bf(e / rs);
    Ps[m * 32 + 16 + (lane & 15)] = 0;           // zero-pad K to 32
  }
  asm volatile("s_wait_dscnt 0" ::: "memory");
  FragBF pf;
  pf.q[0] = *(const uint4*)&Ps[(lane & 15) * 32 + koff];
  pf.q[1] = *(const uint4*)&Ps[(lane & 15) * 32 + 16 + koff];

  #pragma unroll
  for (int nt = 0; nt < 4; ++nt) {
    const int d = nt * 16 + (lane & 15);
    FragBF bv;
    #pragma unroll
    for (int i = 0; i < 8; ++i) {
      const size_t tok = (size_t)(b * Nc + (koff + i) * Pc + p);
      bv.s[i]     = qkv[tok * E3c + vcol + d];
      bv.s[8 + i] = 0;                           // padded K rows
    }
    v8f o = wmma_bf16(pf, bv, v8f_zero());
    #pragma unroll
    for (int r = 0; r < 8; ++r) {
      const int fo = r + mh;
      const size_t tok = (size_t)(b * Nc + fo * Pc + p);
      attn[tok * Dc + gh * 64 + nt * 16 + (lane & 15)] = f2bf(o[r]);
    }
  }
}

// ---------------- launcher ----------------
extern "C" void kernel_launch(void* const* d_in, const int* in_sizes, int n_in,
                              void* d_out, int out_size, void* d_ws, size_t ws_size,
                              hipStream_t stream) {
  const float* x      = (const float*)d_in[0];
  const float* w_qkv  = (const float*)d_in[1];
  const float* w_proj = (const float*)d_in[2];
  const float* b_proj = (const float*)d_in[3];
  float* out = (float*)d_out;

  char* ws = (char*)d_ws;
  uint16_t* xb     = (uint16_t*)(ws + XB_OFF);
  uint16_t* wqkvb  = (uint16_t*)(ws + WQKV_OFF);
  uint16_t* wprojb = (uint16_t*)(ws + WPROJ_OFF);
  uint16_t* qkvb   = (uint16_t*)(ws + QKV_OFF);
  uint16_t* attnb  = (uint16_t*)(ws + ATTN_OFF);

  // 1) f32 -> bf16
  cvt_f32_bf16<<<(Mc * Dc  + 255) / 256, 256, 0, stream>>>(x,      xb,     Mc * Dc);
  cvt_f32_bf16<<<(E3c * Dc + 255) / 256, 256, 0, stream>>>(w_qkv,  wqkvb,  E3c * Dc);
  cvt_f32_bf16<<<(Dc * Dc  + 255) / 256, 256, 0, stream>>>(w_proj, wprojb, Dc * Dc);

  // 2) qkv = x @ w_qkv^T   (25088 x 2304)
  gemm_bf16<false><<<dim3(E3c / 128, Mc / 128), 256, 0, stream>>>(
      xb, wqkvb, qkvb, nullptr, nullptr, Dc, E3c);

  // 3) divided attention -> attnb (25088 x 768, heads interleaved by column)
  attn_spatial <<<Bc * Fc * H2c * 2, 256, 0, stream>>>(qkvb, attnb);
  attn_temporal<<<Bc * Pc * H2c,      32, 0, stream>>>(qkvb, attnb);

  // 4) out = attn @ w_proj^T + b   (f32)
  gemm_bf16<true><<<dim3(Dc / 128, Mc / 128), 256, 0, stream>>>(
      attnb, wprojb, nullptr, out, b_proj, Dc, Dc);
}
